// RNN_arch_2_final_65670049955959
// MI455X (gfx1250) — compile-verified, matrix operand order checked by reading, not run on hardware
//
#include <hip/hip_runtime.h>
#include <hip/hip_bf16.h>
#include <math.h>

// ---------------------------------------------------------------------------
// Elman RNN, T=16, B=8192, hidden=256, in=64, out=64, classes=8.
// Fully fused single kernel: each block owns a 32-row batch tile, keeps all
// weights + h resident in LDS, loops the 16 timesteps. All GEMMs run on
// v_wmma_f32_16x16x32_f16 (f16 in, f32 acc). Per-step x tiles are prefetched
// one step ahead with global_load_async_to_lds_b128 (ASYNCcnt, inline asm),
// and the big streaming outputs use non-temporal stores to protect L2.
// ---------------------------------------------------------------------------

typedef __attribute__((ext_vector_type(16))) _Float16 v16h;
typedef __attribute__((ext_vector_type(8)))  _Float16 v8h;
typedef __attribute__((ext_vector_type(8)))  float    v8f;

#define TT   16
#define BB   8192
#define HID  256
#define IND  64
#define OD   64
#define NCLS 8
#define MT   32          // batch rows per block

// ---- async global -> LDS copy, 16B per lane (CDNA5 §15.18.3, ASYNCcnt) ----
__device__ __forceinline__ void async_cp16(const float* g, float* l) {
  const unsigned int       lds = (unsigned int)(uintptr_t)l;       // LDS byte addr
  const unsigned long long ga  = (unsigned long long)(uintptr_t)g; // global addr
  asm volatile("global_load_async_to_lds_b128 %0, %1, off"
               :: "v"(lds), "v"(ga)
               : "memory");
}

__device__ __forceinline__ void wait_async0() {
  asm volatile("s_wait_asynccnt 0x0" ::: "memory");
}

// A/B fragment load per CDNA5 ISA 7.12.2 (16-bit 16x32 operand):
// lanes 0-15 hold K = {k0..k0+7} U {k0+16..k0+23}; lanes 16-31 are +8.
// Caller passes pointer already offset by (kchunk*32 + ((lane>>4)<<3)).
__device__ __forceinline__ v16h load_frag16(const _Float16* p) {
  v8h lo = *(const v8h*)(p);
  v8h hi = *(const v8h*)(p + 16);
  v16h r;
#pragma unroll
  for (int i = 0; i < 8; ++i) { r[i] = lo[i]; r[i + 8] = hi[i]; }
  return r;
}

__device__ __forceinline__ v8f wmma16(v16h a, v16h b, v8f c) {
  // D = A(16x32 f16) * B(32x16 f16) + C(16x16 f32)
  return __builtin_amdgcn_wmma_f32_16x16x32_f16(false, a, false, b,
                                                (short)0, c, false, false);
}

// ---------------------------------------------------------------------------
// Fused RNN kernel: 256 blocks x 256 threads (8 wave32).
// Per step t:
//   i2h  = x[t] @ Wi^T + bi          (written to i2h_seq)
//   h2h  = h    @ Whh^T + bhh        (written to h2h_seq)
//   h    = tanh(i2h + h2h)           (t==15 -> hc1_final)
//   o    = tanh(h @ Who^T + bho)
//   y    = o @ Wfc^T + bfc           (written to output_seq)
// ---------------------------------------------------------------------------
__global__ __launch_bounds__(256) void rnn_fused_kernel(
    const float* __restrict__ x,     // [T*B, 64]
    const float* __restrict__ hc1,   // [B, 256]
    const float* __restrict__ Wi,    // [256, 64]
    const float* __restrict__ bi,    // [256]
    const float* __restrict__ Whh,   // [256, 256]
    const float* __restrict__ bhh,   // [256]
    const float* __restrict__ Who,   // [64, 256]
    const float* __restrict__ bho,   // [64]
    const float* __restrict__ Wfc,   // [8, 64]
    const float* __restrict__ bfc,   // [8]
    float* __restrict__ out_y,       // [T*B, 8]
    float* __restrict__ out_hc1,     // [B, 256]
    float* __restrict__ out_i2h,     // [T*B, 256]
    float* __restrict__ out_h2h)     // [T*B, 256]
{
  __shared__ _Float16 sW  [HID * HID];   // Whh f16          (128 KB)
  __shared__ _Float16 sWi [HID * IND];   // Wi  f16          ( 32 KB)
  __shared__ _Float16 sWo [OD * HID];    // Who f16          ( 32 KB)
  __shared__ _Float16 sH  [MT * HID];    // h   f16          ( 16 KB)
  __shared__ float    sXr [MT * IND];    // raw x tile (t+1) (  8 KB)
  __shared__ _Float16 sXh [MT * IND];    // f16 x tile (t)   (  4 KB)
  __shared__ _Float16 sO  [MT * OD];     // o   f16          (  4 KB)
  __shared__ _Float16 sWf [16 * OD];     // Wfc padded 8->16 (  2 KB)
  __shared__ float sBh[HID];
  __shared__ float sBi[HID];
  __shared__ float sBo[OD];
  __shared__ float sBf[NCLS];

  const int tid  = threadIdx.x;
  const int lane = tid & 31;
  const int wave = tid >> 5;
  const size_t rowBase = (size_t)blockIdx.x * MT;

  // ---- stage weights / initial state / x tile 0 into LDS ----
#pragma unroll 4
  for (int i = 0; i < 256; ++i)                        // Whh: 65536
    sW[tid + i * 256] = (_Float16)Whh[tid + i * 256];
#pragma unroll 4
  for (int i = 0; i < 64; ++i)                         // Wi: 16384
    sWi[tid + i * 256] = (_Float16)Wi[tid + i * 256];
#pragma unroll 4
  for (int i = 0; i < 64; ++i)                         // Who: 16384
    sWo[tid + i * 256] = (_Float16)Who[tid + i * 256];
#pragma unroll
  for (int i = 0; i < 32; ++i)                         // hc1 tile: 8192
    sH[tid + i * 256] = (_Float16)hc1[rowBase * HID + tid + i * 256];
#pragma unroll
  for (int i = 0; i < 8; ++i) {                        // x tile t=0: 2048
    const int idx = tid + i * 256;
    sXh[idx] = (_Float16)x[rowBase * IND + idx];
  }
  for (int idx = tid; idx < 16 * OD; idx += 256) {     // Wfc padded rows
    const int r = idx >> 6;
    sWf[idx] = (r < NCLS) ? (_Float16)Wfc[idx] : (_Float16)0.0f;
  }
  sBh[tid] = bhh[tid];
  sBi[tid] = bi[tid];
  if (tid < OD)   sBo[tid] = bho[tid];
  if (tid < NCLS) sBf[tid] = bfc[tid];
  __syncthreads();

  const int mr  = wave >> 2;             // row tile 0..1
  const int nb  = (wave & 3) * 64;       // 4 col tiles per wave (h2h / i2h)
  const int nbO = (wave & 3) * 16;       // 1 col tile per wave  (out head)
  const int hi8 = (lane >> 4) << 3;
  const int l15 = lane & 15;
  const v8f vzero = {};

  for (int t = 0; t < TT; ++t) {
    const size_t tRow = (size_t)t * BB + rowBase;

    // ---- issue async prefetch of raw x tile (t+1) into sXr ----
    if (t + 1 < TT) {
      const float* xg = x + ((size_t)(t + 1) * BB + rowBase) * IND;
#pragma unroll
      for (int j = 0; j < 2; ++j) {      // 2048 floats, 16B per lane-op
        const int e = (tid + j * 256) * 4;
        async_cp16(xg + e, sXr + e);
      }
    }

    // ---- h2h: [32,256] = sH x sW^T, K=256 ----
    v8f acc[4];
#pragma unroll
    for (int i = 0; i < 4; ++i) acc[i] = vzero;
#pragma unroll 2
    for (int kc = 0; kc < 8; ++kc) {
      const int k0 = kc * 32 + hi8;
      v16h a = load_frag16(&sH[(mr * 16 + l15) * HID + k0]);
#pragma unroll
      for (int n = 0; n < 4; ++n) {
        v16h b = load_frag16(&sW[(nb + n * 16 + l15) * HID + k0]);
        acc[n] = wmma16(a, b, acc[n]);
      }
    }

    // ---- i2h: [32,256] = sXh x sWi^T, K=64 ----
    v8f ac2[4];
#pragma unroll
    for (int i = 0; i < 4; ++i) ac2[i] = vzero;
#pragma unroll
    for (int kc = 0; kc < 2; ++kc) {
      const int k0 = kc * 32 + hi8;
      v16h a = load_frag16(&sXh[(mr * 16 + l15) * IND + k0]);
#pragma unroll
      for (int n = 0; n < 4; ++n) {
        v16h b = load_frag16(&sWi[(nb + n * 16 + l15) * IND + k0]);
        ac2[n] = wmma16(a, b, ac2[n]);
      }
    }

    wait_async0();       // sXr(t+1) complete (this wave's issues)
    __syncthreads();     // + all waves done reading sH / sXh

    // ---- epilogue: write i2h_seq & h2h_seq, h = tanh(i2h + h2h) ----
#pragma unroll
    for (int n = 0; n < 4; ++n) {
      const int col = nb + n * 16 + l15;
      const float bhv = sBh[col];
      const float biv = sBi[col];
#pragma unroll
      for (int v = 0; v < 8; ++v) {
        const int m = mr * 16 + v + hi8;
        const float i2hv = ac2[n][v] + biv;
        const float h2hv = acc[n][v] + bhv;
        const size_t g = (tRow + m) * HID + col;
        __builtin_nontemporal_store(i2hv, &out_i2h[g]);
        __builtin_nontemporal_store(h2hv, &out_h2h[g]);
        const float hn = tanhf(i2hv + h2hv);
        sH[m * HID + col] = (_Float16)hn;
        if (t == TT - 1) out_hc1[(rowBase + m) * HID + col] = hn;
      }
    }
    // convert raw x tile (t+1) -> f16 for next step
    if (t + 1 < TT) {
#pragma unroll
      for (int j = 0; j < 8; ++j) {
        const int idx = tid + j * 256;
        sXh[idx] = (_Float16)sXr[idx];
      }
    }
    __syncthreads();     // new sH / sXh visible

    // ---- out head: o = tanh(sH x sWo^T + bho), M=32 N=64 K=256 ----
    {
      v8f ac3 = vzero;
#pragma unroll 2
      for (int kc = 0; kc < 8; ++kc) {
        const int k0 = kc * 32 + hi8;
        v16h a = load_frag16(&sH[(mr * 16 + l15) * HID + k0]);
        v16h b = load_frag16(&sWo[(nbO + l15) * HID + k0]);
        ac3 = wmma16(a, b, ac3);
      }
      const int col = nbO + l15;
      const float bv = sBo[col];
#pragma unroll
      for (int v = 0; v < 8; ++v) {
        const int m = mr * 16 + v + hi8;
        sO[m * OD + col] = (_Float16)tanhf(ac3[v] + bv);
      }
    }
    __syncthreads();     // sO ready

    // ---- classifier: y = sO x sWf^T + bfc, M=32 N=16(pad 8) K=64 ----
    if (wave < 2) {
      v8f ac4 = vzero;
#pragma unroll
      for (int kc = 0; kc < 2; ++kc) {
        const int k0 = kc * 32 + hi8;
        v16h a = load_frag16(&sO[(wave * 16 + l15) * OD + k0]);
        v16h b = load_frag16(&sWf[l15 * OD + k0]);
        ac4 = wmma16(a, b, ac4);
      }
      if (l15 < NCLS) {
        const float bv = sBf[l15];
#pragma unroll
        for (int v = 0; v < 8; ++v) {
          const int m = wave * 16 + v + hi8;
          out_y[(tRow + m) * NCLS + l15] = ac4[v] + bv;
        }
      }
    }
    __syncthreads();     // protect sO / sH / sXr for next step
  }
}

// ---------------------------------------------------------------------------
// Output layout (flat f32, reference return order):
//   output_seq [16,8192,8]    @ 0          (1048576)
//   hc1_final  [8192,256]     @ 1048576    (2097152)
//   hc2        [8192,256]     @ 3145728
//   hc3        [8192,256]     @ 5242880
//   hc4        [8192,256]     @ 7340032
//   i2h_seq    [16,8192,256]  @ 9437184    (33554432)
//   h2h_seq    [16,8192,256]  @ 42991616   (33554432)
// ---------------------------------------------------------------------------
extern "C" void kernel_launch(void* const* d_in, const int* in_sizes, int n_in,
                              void* d_out, int out_size, void* d_ws, size_t ws_size,
                              hipStream_t stream) {
  (void)in_sizes; (void)n_in; (void)out_size; (void)d_ws; (void)ws_size;
  const float* x    = (const float*)d_in[0];
  /* cue = d_in[1] unused by reference */
  const float* hc1  = (const float*)d_in[2];
  const float* hc2  = (const float*)d_in[3];
  const float* hc3  = (const float*)d_in[4];
  const float* hc4  = (const float*)d_in[5];
  const float* Wi2h = (const float*)d_in[6];
  const float* bi2h = (const float*)d_in[7];
  const float* Whh  = (const float*)d_in[8];
  const float* bhh  = (const float*)d_in[9];
  const float* Who  = (const float*)d_in[10];
  const float* bho  = (const float*)d_in[11];
  const float* Wfc  = (const float*)d_in[12];
  const float* bfc  = (const float*)d_in[13];

  float* out      = (float*)d_out;
  float* out_y    = out + 0;
  float* out_hc1  = out + 1048576;
  float* out_hc2  = out + 3145728;
  float* out_hc3  = out + 5242880;
  float* out_hc4  = out + 7340032;
  float* out_i2h  = out + 9437184;
  float* out_h2h  = out + 42991616;

  const size_t hbytes = (size_t)BB * HID * sizeof(float);
  (void)hipMemcpyAsync(out_hc2, hc2, hbytes, hipMemcpyDeviceToDevice, stream);
  (void)hipMemcpyAsync(out_hc3, hc3, hbytes, hipMemcpyDeviceToDevice, stream);
  (void)hipMemcpyAsync(out_hc4, hc4, hbytes, hipMemcpyDeviceToDevice, stream);

  // 8192 batch rows / 32 per block = 256 blocks, 8 wave32 each
  rnn_fused_kernel<<<BB / MT, 256, 0, stream>>>(
      x, hc1, Wi2h, bi2h, Whh, bhh, Who, bho, Wfc, bfc,
      out_y, out_hc1, out_i2h, out_h2h);
}
